// NN_36541581754944
// MI455X (gfx1250) — matrix operand order repaired and checked by reading, not compiled, not run on hardware
//
#include <hip/hip_runtime.h>

#define K_DIM    40960
#define KT_COUNT 1280      // K_DIM / 32
#define ACCH     256
#define NT       16        // ACCH / 16 output tiles per wave (full N)
#define BATCH    8192
#define HID      32

typedef __attribute__((ext_vector_type(16))) __bf16 v16bf;
typedef __attribute__((ext_vector_type(8)))  float  v8f;
typedef __attribute__((ext_vector_type(4)))  float  f4;

static __device__ __forceinline__ unsigned short f2bf(float f) {
    unsigned int x = __builtin_bit_cast(unsigned int, f);
    x += 0x7FFFu + ((x >> 16) & 1u);          // round-to-nearest-even
    return (unsigned short)(x >> 16);
}

static __device__ __forceinline__ float clip01(float x) {
    return fminf(fmaxf(x, 0.0f), 1.0f);
}

// ---------------------------------------------------------------------------
// Pack w_white / w_black (fp32 [256, 40960] row-major) into bf16 B-fragment
// order: [color][ktile][ntile][lane][16]. Lane layout, contiguous k order (16-bit B,
// 32x16): lanes 0-15 -> N=lane, K = kt*32 + 0..15 ; lanes 16-31 -> N=lane-16,
// K = kt*32 + 16..31. Each lane's 16 values are contiguous in memory.
// ---------------------------------------------------------------------------
__global__ __launch_bounds__(256) void nnue_pack_w(
        const float* __restrict__ wW, const float* __restrict__ wB,
        unsigned short* __restrict__ packed) {
    int tid   = blockIdx.x * 256 + threadIdx.x;   // [0, 2*1280*16*32)
    int lane  = tid & 31;
    int ntg   = (tid >> 5) & 15;
    int kt    = (tid >> 9) % KT_COUNT;
    int color = tid / (KT_COUNT * NT * 32);

    const float* w = color ? wB : wW;
    int n      = ntg * 16 + (lane & 15);
    int kstart = kt * 32 + ((lane >> 4) << 4);
    const float* src = w + (size_t)n * K_DIM + kstart;

    unsigned short tmp[16];
    #pragma unroll
    for (int j = 0; j < 16; j += 4) {
        f4 v = *(const f4*)(src + j);
        tmp[j + 0] = f2bf(v.x); tmp[j + 1] = f2bf(v.y);
        tmp[j + 2] = f2bf(v.z); tmp[j + 3] = f2bf(v.w);
    }
    unsigned short* dst = packed +
        ((size_t)color * KT_COUNT * NT * 32 + ((size_t)kt * NT + ntg) * 32 + lane) * 16;
    ((uint4*)dst)[0] = *(const uint4*)&tmp[0];
    ((uint4*)dst)[1] = *(const uint4*)&tmp[8];
}

// ---------------------------------------------------------------------------
// Feature-transformer GEMM: acc[color] = feats[color] @ w[color]^T + b[color]
// grid = (64 m-blocks, 2 colors), 256 threads = 8 waves.
// Wave: 16 rows x FULL 256 cols (16 WMMA tiles, 128 acc VGPRs), so each
// activation row is streamed from HBM exactly once; weight reuse comes from
// the bf16-packed tile living in L2 (40 MB << 192 MB).
// ---------------------------------------------------------------------------
__global__ __launch_bounds__(256, 1) void nnue_ft_gemm(
        const float* __restrict__ whiteF, const float* __restrict__ blackF,
        const float* __restrict__ bW, const float* __restrict__ bB,
        const unsigned short* __restrict__ packedW,
        float* __restrict__ whiteAcc, float* __restrict__ blackAcc) {
    const int lane   = threadIdx.x & 31;
    const int wave   = threadIdx.x >> 5;
    const int mblock = blockIdx.x;            // 0..63
    const int color  = blockIdx.y;            // 0..1

    const float* A    = color ? blackF : whiteF;
    const float* bias = color ? bB : bW;
    float*       C    = color ? blackAcc : whiteAcc;
    const unsigned short* wp = packedW + (size_t)color * KT_COUNT * NT * 32 * 16;

    const int mrow = mblock * 128 + wave * 16 + (lane & 15);
    const int half = lane >> 4;
    const float* arow = A + (size_t)mrow * K_DIM;

    v8f acc[NT] = {};

    for (int kt = 0; kt < KT_COUNT; ++kt) {
        // A fragment: contiguous K order matching packed B. elems 0..15 =
        // K kbase+half*16+{0..15}
        const float* ap = arow + kt * 32 + half * 16;
        __builtin_prefetch(ap + 512, 0, 1);            // ~16 k-tiles ahead
        f4 a0 = *(const f4*)(ap);
        f4 a1 = *(const f4*)(ap + 4);
        f4 a2 = *(const f4*)(ap + 8);
        f4 a3 = *(const f4*)(ap + 12);
        union { v16bf v; unsigned short u[16]; } af;
        #pragma unroll
        for (int j = 0; j < 4; ++j) {
            af.u[j]      = f2bf(a0[j]);
            af.u[4 + j]  = f2bf(a1[j]);
            af.u[8 + j]  = f2bf(a2[j]);
            af.u[12 + j] = f2bf(a3[j]);
        }
        const unsigned short* wk = wp + (((size_t)kt * NT) * 32 + lane) * 16;
        #pragma unroll
        for (int n = 0; n < NT; ++n) {
            v16bf bfrag = *(const v16bf*)(wk + (size_t)n * (32 * 16));
            acc[n] = __builtin_amdgcn_wmma_f32_16x16x32_bf16(
                false, af.v, false, bfrag, (short)0, acc[n], false, false);
        }
    }

    // Epilogue: C/D layout -> lanes 0-15: M = r, lanes 16-31: M = 8 + r; N = lane&15
    const int mbase = mblock * 128 + wave * 16 + half * 8;
    #pragma unroll
    for (int n = 0; n < NT; ++n) {
        int c = n * 16 + (lane & 15);
        float bv = bias[c];
        float* cp = C + (size_t)mbase * ACCH + c;
        #pragma unroll
        for (int r = 0; r < 8; ++r)
            cp[(size_t)r * ACCH] = acc[n][r] + bv;
    }
}

// ---------------------------------------------------------------------------
// Tail MLP: one wave per row. Lane l = hidden neuron l (HID == 32 == wave32).
// ---------------------------------------------------------------------------
__global__ __launch_bounds__(256) void nnue_tail(
        const float* __restrict__ whiteAcc, const float* __restrict__ blackAcc,
        const float* __restrict__ stm,
        const float* __restrict__ w1, const float* __restrict__ b1,
        const float* __restrict__ w2, const float* __restrict__ b2,
        const float* __restrict__ wOut, const float* __restrict__ bOut,
        float* __restrict__ out) {
    __shared__ float w1t[512 * HID];   // transposed: w1t[k*32 + o] = w1[o*512 + k]
    for (int idx = threadIdx.x; idx < HID * 512; idx += 256) {
        int o = idx >> 9, k = idx & 511;
        w1t[k * HID + o] = w1[idx];
    }
    __syncthreads();

    const int lane = threadIdx.x & 31;
    const int wave = threadIdx.x >> 5;
    const int row  = blockIdx.x * 8 + wave;

    const float s  = stm[row];
    const float* wr = whiteAcc + (size_t)row * ACCH;
    const float* br = blackAcc + (size_t)row * ACCH;

    float sum = b1[lane];
    for (int k = 0; k < 256; ++k) {
        float wv = wr[k], bv = br[k];
        float a0 = clip01(wv + s * (bv - wv));     // concat col k
        float a1 = clip01(bv + s * (wv - bv));     // concat col 256+k
        sum += a0 * w1t[k * HID + lane];
        sum += a1 * w1t[(256 + k) * HID + lane];
    }
    float in2  = clip01(sum);
    float sum2 = b2[lane];
    #pragma unroll
    for (int j = 0; j < HID; ++j)
        sum2 += __shfl(in2, j, 32) * w2[lane * HID + j];

    float p = clip01(sum2) * wOut[lane];
    #pragma unroll
    for (int off = 16; off; off >>= 1) p += __shfl_xor(p, off, 32);
    if (lane == 0) out[row] = p + bOut[0];
}

// ---------------------------------------------------------------------------
extern "C" void kernel_launch(void* const* d_in, const int* in_sizes, int n_in,
                              void* d_out, int out_size, void* d_ws, size_t ws_size,
                              hipStream_t stream) {
    const float* whiteF = (const float*)d_in[0];
    const float* blackF = (const float*)d_in[1];
    const float* stm    = (const float*)d_in[2];
    const float* wW     = (const float*)d_in[3];
    const float* bW     = (const float*)d_in[4];
    const float* wB     = (const float*)d_in[5];
    const float* bB     = (const float*)d_in[6];
    const float* w1     = (const float*)d_in[7];
    const float* b1     = (const float*)d_in[8];
    const float* w2     = (const float*)d_in[9];
    const float* b2     = (const float*)d_in[10];
    const float* wOut   = (const float*)d_in[11];
    const float* bOut   = (const float*)d_in[12];
    float* out = (float*)d_out;

    unsigned char* ws = (unsigned char*)d_ws;
    unsigned short* packed = (unsigned short*)ws;                      // 40 MB
    size_t packedBytes = (size_t)2 * KT_COUNT * NT * 32 * 16 * sizeof(unsigned short);
    float* whiteAcc = (float*)(ws + packedBytes);                      // 8 MB
    float* blackAcc = whiteAcc + (size_t)BATCH * ACCH;                 // 8 MB

    int packTotal = 2 * KT_COUNT * NT * 32;
    nnue_pack_w<<<packTotal / 256, 256, 0, stream>>>(wW, wB, packed);

    dim3 g(BATCH / 128, 2);
    nnue_ft_gemm<<<g, 256, 0, stream>>>(whiteF, blackF, bW, bB, packed,
                                        whiteAcc, blackAcc);

    nnue_tail<<<BATCH / 8, 256, 0, stream>>>(whiteAcc, blackAcc, stm,
                                             w1, b1, w2, b2, wOut, bOut, out);
}
